// self_attention_64604898067011
// MI455X (gfx1250) — matrix operand order repaired and checked by reading, not compiled, not run on hardware
//
#include <hip/hip_runtime.h>
#include <hip/hip_bf16.h>

// ---------------------------------------------------------------------------
// MSDA (multi-scale deformable attention, 1 level) for MI455X / gfx1250.
// bs=8, nq=4096 (64x64), E=256, NH=8, hd=32, NP=4.  M = bs*nq = 32768.
//
// Pipeline (stream-ordered, graph-capture safe):
//   prep      : weights fp32 -> bf16 (concat [value_w;off_w;attn_w] = 352x256)
//   proj_gemm : C[M,352] = Q[M,256] x Wcat^T  via v_wmma_f32_16x16x32_bf16
//   sample    : softmax(4) + bilinear gather-weighted sum  -> sampled[M,256]
//   out_gemm  : out = sampled x out_w^T + out_b + 2*query  via WMMA
//
// GEMM scheme: block = 256 thr (8 waves), M_TILE = 32 rows staged to LDS as
// bf16.  Each wave owns one 16-wide N-tile at a time; B fragments are
// software-pipelined (ping-pong, next load issued before current WMMAs) so a
// load pair is always in flight behind the 2-WMMA chain.  Accumulator tiles
// are transposed through wave-private LDS scratch and written as b128 stores.
// ---------------------------------------------------------------------------

typedef __attribute__((ext_vector_type(16))) __bf16 v16bf;
typedef __attribute__((ext_vector_type(8)))  __bf16 v8bf;
typedef __attribute__((ext_vector_type(4)))  __bf16 v4bf;
typedef __attribute__((ext_vector_type(8)))  float  v8f;

#define M_TOTAL 32768   // bs*nq
#define E_DIM   256
#define NCAT    352     // 256 (value) + 64 (off) + 32 (attn)
#define GRID_HW 64
#define M_TILE  32
#define CS_PITCH 20     // 16 + 4 pad (floats)

// ---------------------------------------------------------------------------
// Kernel 0: weight conversion / concat to bf16.
// ---------------------------------------------------------------------------
__global__ __launch_bounds__(256) void msda_prep(
    const float* __restrict__ value_w, const float* __restrict__ off_w,
    const float* __restrict__ attn_w, const float* __restrict__ out_w,
    __bf16* __restrict__ wcat, __bf16* __restrict__ outw)
{
    int t = blockIdx.x * 256 + threadIdx.x;
    if (t < NCAT * E_DIM) {
        int n = t >> 8, k = t & 255;
        float s = (n < 256) ? value_w[n * 256 + k]
                : (n < 320) ? off_w[(n - 256) * 256 + k]
                            : attn_w[(n - 320) * 256 + k];
        wcat[t] = (__bf16)s;
    }
    if (t < E_DIM * E_DIM) outw[t] = (__bf16)out_w[t];
}

// ---------------------------------------------------------------------------
// WMMA fragment helpers (wave32, 16-bit A/B in contiguous k order used by this code)
//   A 16x32: lanes 0-15 row M=lane, K = 0..15; lanes 16-31 K = 16..31
//   B 32x16: lanes 0-15 col N=lane, K = 0..15; lanes 16-31 K = 16..31
//   C 16x16: vgpr r -> row r (+8 for upper half-wave), col = lane&15
// ---------------------------------------------------------------------------
__device__ inline v16bf load_a_frag(const __bf16* rowp, int k0, int subA)
{
    v8bf lo = *(const v8bf*)(rowp + k0 + 2 * subA);
    v8bf hi = *(const v8bf*)(rowp + k0 + 2 * subA + 8);
    return __builtin_shufflevector(lo, hi, 0, 1, 2, 3, 4, 5, 6, 7,
                                           8, 9, 10, 11, 12, 13, 14, 15);
}

// Stage M_TILE x 256 fp32 rows into LDS as bf16 (packed v4bf stores).
__device__ inline void stage_tile_bf16(const float* __restrict__ src,
                                       __bf16 (*As)[264], int tid)
{
    const float4* s4 = (const float4*)src;
    #pragma unroll
    for (int i = 0; i < (M_TILE * 64) / 256; ++i) {   // 8 iters
        int e   = tid + i * 256;
        int row = e >> 6;            // 64 float4 per row
        int c   = (e & 63) * 4;
        float4 v = s4[e];
        v4bf p = { (__bf16)v.x, (__bf16)v.y, (__bf16)v.z, (__bf16)v.w };
        *(v4bf*)&As[row][c] = p;
    }
}

// 2-tile (32x16) WMMA chain with ping-pong B pipeline.
__device__ inline void wmma_chain(const v16bf* afrag0, const v16bf* afrag1,
                                  const __bf16* brow, v8f& acc0, v8f& acc1)
{
    v16bf bcur = *(const v16bf*)(brow);
    #pragma unroll
    for (int kk = 0; kk < 8; ++kk) {
        v16bf bnxt = bcur;
        if (kk < 7) bnxt = *(const v16bf*)(brow + (kk + 1) * 32);
        acc0 = __builtin_amdgcn_wmma_f32_16x16x32_bf16(
            false, afrag0[kk], false, bcur, (short)0, acc0, false, false);
        acc1 = __builtin_amdgcn_wmma_f32_16x16x32_bf16(
            false, afrag1[kk], false, bcur, (short)0, acc1, false, false);
        bcur = bnxt;
    }
}

// ---------------------------------------------------------------------------
// Kernel 1: fused projection GEMM.
// ---------------------------------------------------------------------------
__global__ __launch_bounds__(256) void msda_proj_gemm(
    const float* __restrict__ query, const __bf16* __restrict__ wcat,
    const float* __restrict__ value_b, const float* __restrict__ off_b,
    const float* __restrict__ attn_b,
    float* __restrict__ value, float* __restrict__ off, float* __restrict__ attn)
{
    __shared__ __align__(16) __bf16 As[M_TILE][264];       // 32x256 bf16 (+pad)
    __shared__ __align__(16) float  Cs[8][16][CS_PITCH];   // wave-private C tile

    const int m0  = blockIdx.x * M_TILE;
    const int tid = threadIdx.x;

    stage_tile_bf16(query + (size_t)m0 * E_DIM, As, tid);
    __syncthreads();

    const int lane = tid & 31;
    const int wave = tid >> 5;
    const int nl   = lane & 15;
    const int half = lane >> 4;
    const int subA = half * 8;

    // A fragments for both 16-row subtiles, register resident.
    v16bf afrag0[8], afrag1[8];
    #pragma unroll
    for (int kk = 0; kk < 8; ++kk) {
        afrag0[kk] = load_a_frag(&As[nl][0],      kk * 32, subA);
        afrag1[kk] = load_a_frag(&As[16 + nl][0], kk * 32, subA);
    }

    float* cs = &Cs[wave][0][0];
    const int rrow = lane >> 1;            // 0..15 (row within subtile)
    const int c0   = (lane & 1) * 8;       // 0 or 8 (col group)

    for (int t = wave; t < NCAT / 16; t += 8) {
        const int n = t * 16 + nl;                       // output column
        const __bf16* brow = wcat + (size_t)n * E_DIM + half * 16;

        v8f acc0 = {}, acc1 = {};
        wmma_chain(afrag0, afrag1, brow, acc0, acc1);

        // Region routing is uniform per N-tile (boundaries 256/320 are
        // multiples of 16) -> scalar branch, no exec-mask churn.
        float* base;
        int    pitch, nbase;
        float  bias;
        if (t < 16)      { base = value; pitch = 256; nbase = t * 16;       bias = value_b[n];       }
        else if (t < 20) { base = off;   pitch = 64;  nbase = t * 16 - 256; bias = off_b[n - 256];   }
        else             { base = attn;  pitch = 32;  nbase = t * 16 - 320; bias = attn_b[n - 320];  }

        // Subtile 0: acc -> LDS (bias fused) -> row-major b128 stores.
        #pragma unroll
        for (int r = 0; r < 8; ++r)
            cs[(r + half * 8) * CS_PITCH + nl] = acc0[r] + bias;
        {
            float4 u0 = *(float4*)&cs[rrow * CS_PITCH + c0];
            float4 u1 = *(float4*)&cs[rrow * CS_PITCH + c0 + 4];
            float* dst = base + (size_t)(m0 + rrow) * pitch + nbase + c0;
            *(float4*)(dst)     = u0;
            *(float4*)(dst + 4) = u1;
        }
        // Subtile 1 (rows +16).
        #pragma unroll
        for (int r = 0; r < 8; ++r)
            cs[(r + half * 8) * CS_PITCH + nl] = acc1[r] + bias;
        {
            float4 u0 = *(float4*)&cs[rrow * CS_PITCH + c0];
            float4 u1 = *(float4*)&cs[rrow * CS_PITCH + c0 + 4];
            float* dst = base + (size_t)(m0 + 16 + rrow) * pitch + nbase + c0;
            *(float4*)(dst)     = u0;
            *(float4*)(dst + 4) = u1;
        }
    }
}

// ---------------------------------------------------------------------------
// Kernel 2: softmax(NP=4) + bilinear sampling.  8 threads per (b,h,q),
// each owns one float4 chunk of hd=32.  g = (b*NH+h)*nq + q keeps
// neighboring q together -> spatially local gathers.
// ---------------------------------------------------------------------------
__global__ __launch_bounds__(256) void msda_sample(
    const float* __restrict__ value, const float* __restrict__ off,
    const float* __restrict__ attn, float* __restrict__ sampled)
{
    int t     = blockIdx.x * 256 + threadIdx.x;   // 2,097,152 threads
    int chunk = t & 7;
    int g     = t >> 3;
    int q     = g & 4095;
    int bh    = g >> 12;
    int b     = bh >> 3;
    int h     = bh & 7;

    int row = q >> 6, col = q & 63;
    float refx = (float)col * (1.0f / 63.0f);
    float refy = (float)row * (1.0f / 63.0f);

    const float* offp = off  + ((size_t)(b * 4096 + q)) * 64 + h * 8;
    const float* attp = attn + ((size_t)(b * 4096 + q)) * 32 + h * 4;

    float l0 = attp[0], l1 = attp[1], l2 = attp[2], l3 = attp[3];
    float mx = fmaxf(fmaxf(l0, l1), fmaxf(l2, l3));
    float e0 = __expf(l0 - mx), e1 = __expf(l1 - mx);
    float e2 = __expf(l2 - mx), e3 = __expf(l3 - mx);
    float inv = 1.0f / (e0 + e1 + e2 + e3);
    float aw[4] = { e0 * inv, e1 * inv, e2 * inv, e3 * inv };

    const float4* v4 = (const float4*)value;
    float4 acc = make_float4(0.f, 0.f, 0.f, 0.f);

    #pragma unroll
    for (int p = 0; p < 4; ++p) {
        float px = (refx + offp[p * 2 + 0] * (1.0f / GRID_HW)) * GRID_HW - 0.5f;
        float py = (refy + offp[p * 2 + 1] * (1.0f / GRID_HW)) * GRID_HW - 0.5f;
        float fx = floorf(px), fy = floorf(py);
        int   x0 = (int)fx,    y0 = (int)fy;
        float wx = px - fx,    wy = py - fy;

        #pragma unroll
        for (int c = 0; c < 4; ++c) {
            int dx = c & 1, dy = c >> 1;
            int xi = x0 + dx, yi = y0 + dy;
            float w = (dx ? wx : 1.f - wx) * (dy ? wy : 1.f - wy) * aw[p];
            if (xi >= 0 && xi < GRID_HW && yi >= 0 && yi < GRID_HW) {
                int pos = yi * GRID_HW + xi;
                float4 gv = v4[(size_t)(b * 4096 + pos) * 64 + h * 8 + chunk];
                acc.x += w * gv.x; acc.y += w * gv.y;
                acc.z += w * gv.z; acc.w += w * gv.w;
            }
        }
    }
    ((float4*)sampled)[(size_t)(b * 4096 + q) * 64 + h * 8 + chunk] = acc;
}

// ---------------------------------------------------------------------------
// Kernel 3: output GEMM + bias + 2*query residual.
// ---------------------------------------------------------------------------
__global__ __launch_bounds__(256) void msda_out_gemm(
    const float* __restrict__ sampled, const __bf16* __restrict__ outw,
    const float* __restrict__ out_b, const float* __restrict__ query,
    float* __restrict__ out)
{
    __shared__ __align__(16) __bf16 As[M_TILE][264];
    __shared__ __align__(16) float  Cs[8][16][CS_PITCH];

    const int m0  = blockIdx.x * M_TILE;
    const int tid = threadIdx.x;

    stage_tile_bf16(sampled + (size_t)m0 * E_DIM, As, tid);
    __syncthreads();

    const int lane = tid & 31;
    const int wave = tid >> 5;
    const int nl   = lane & 15;
    const int half = lane >> 4;
    const int subA = half * 8;

    v16bf afrag0[8], afrag1[8];
    #pragma unroll
    for (int kk = 0; kk < 8; ++kk) {
        afrag0[kk] = load_a_frag(&As[nl][0],      kk * 32, subA);
        afrag1[kk] = load_a_frag(&As[16 + nl][0], kk * 32, subA);
    }

    float* cs = &Cs[wave][0][0];
    const int rrow = lane >> 1;
    const int c0   = (lane & 1) * 8;

    #pragma unroll
    for (int t = wave; t < 16; t += 8) {
        const int n = t * 16 + nl;
        const int nbase = t * 16;
        const __bf16* brow = outw + (size_t)n * E_DIM + half * 16;

        v8f acc0 = {}, acc1 = {};
        wmma_chain(afrag0, afrag1, brow, acc0, acc1);

        float bias = out_b[n];

        #pragma unroll
        for (int r = 0; r < 8; ++r)
            cs[(r + half * 8) * CS_PITCH + nl] = acc0[r] + bias;
        {
            float4 u0 = *(float4*)&cs[rrow * CS_PITCH + c0];
            float4 u1 = *(float4*)&cs[rrow * CS_PITCH + c0 + 4];
            size_t idx = (size_t)(m0 + rrow) * 256 + nbase + c0;
            float4 q0 = *(const float4*)&query[idx];
            float4 q1 = *(const float4*)&query[idx + 4];
            float4 o0 = { u0.x + 2.f * q0.x, u0.y + 2.f * q0.y,
                          u0.z + 2.f * q0.z, u0.w + 2.f * q0.w };
            float4 o1 = { u1.x + 2.f * q1.x, u1.y + 2.f * q1.y,
                          u1.z + 2.f * q1.z, u1.w + 2.f * q1.w };
            *(float4*)&out[idx]     = o0;
            *(float4*)&out[idx + 4] = o1;
        }
        #pragma unroll
        for (int r = 0; r < 8; ++r)
            cs[(r + half * 8) * CS_PITCH + nl] = acc1[r] + bias;
        {
            float4 u0 = *(float4*)&cs[rrow * CS_PITCH + c0];
            float4 u1 = *(float4*)&cs[rrow * CS_PITCH + c0 + 4];
            size_t idx = (size_t)(m0 + 16 + rrow) * 256 + nbase + c0;
            float4 q0 = *(const float4*)&query[idx];
            float4 q1 = *(const float4*)&query[idx + 4];
            float4 o0 = { u0.x + 2.f * q0.x, u0.y + 2.f * q0.y,
                          u0.z + 2.f * q0.z, u0.w + 2.f * q0.w };
            float4 o1 = { u1.x + 2.f * q1.x, u1.y + 2.f * q1.y,
                          u1.z + 2.f * q1.z, u1.w + 2.f * q1.w };
            *(float4*)&out[idx]     = o0;
            *(float4*)&out[idx + 4] = o1;
        }
    }
}

// ---------------------------------------------------------------------------
extern "C" void kernel_launch(void* const* d_in, const int* in_sizes, int n_in,
                              void* d_out, int out_size, void* d_ws, size_t ws_size,
                              hipStream_t stream)
{
    const float* query   = (const float*)d_in[0];
    const float* value_w = (const float*)d_in[1];
    const float* value_b = (const float*)d_in[2];
    const float* off_w   = (const float*)d_in[3];
    const float* off_b   = (const float*)d_in[4];
    const float* attn_w  = (const float*)d_in[5];
    const float* attn_b  = (const float*)d_in[6];
    const float* out_w   = (const float*)d_in[7];
    const float* out_b   = (const float*)d_in[8];
    float* out = (float*)d_out;

    // Workspace layout (floats): value | off | attn | sampled | bf16 weights
    const size_t N_VALUE = (size_t)M_TOTAL * 256;  // 8,388,608
    const size_t N_OFF   = (size_t)M_TOTAL * 64;   // 2,097,152
    const size_t N_ATT   = (size_t)M_TOTAL * 32;   // 1,048,576

    float*  value   = (float*)d_ws;
    float*  off     = value + N_VALUE;
    float*  attn    = off + N_OFF;
    float*  sampled = attn + N_ATT;
    __bf16* wcat    = (__bf16*)(sampled + N_VALUE);
    __bf16* outw    = wcat + (size_t)NCAT * E_DIM;

    msda_prep<<<(NCAT * E_DIM + 255) / 256, 256, 0, stream>>>(
        value_w, off_w, attn_w, out_w, wcat, outw);

    msda_proj_gemm<<<M_TOTAL / M_TILE, 256, 0, stream>>>(
        query, wcat, value_b, off_b, attn_b, value, off, attn);

    msda_sample<<<(M_TOTAL * 8 * 8) / 256, 256, 0, stream>>>(
        value, off, attn, sampled);

    msda_out_gemm<<<M_TOTAL / M_TILE, 256, 0, stream>>>(
        sampled, outw, out_b, query, out);
}
